// Qwen2MoeSparseMoeBlock_40286793237171
// MI455X (gfx1250) — matrix operand order repaired and checked by reading, not compiled, not run on hardware
//
#include <hip/hip_runtime.h>
#include <hip/hip_bf16.h>

// Problem constants
#define BB 2
#define SS 1024
#define DD 2048
#define FF 1408
#define FSS 5632
#define EE 16
#define KK 4
#define NTOK (BB * SS)          // 2048 tokens
#define OUT_FINAL ((size_t)NTOK * DD)

typedef __attribute__((ext_vector_type(16))) __bf16 v16bf;
typedef __attribute__((ext_vector_type(8)))  float  v8f;

__device__ __forceinline__ __bf16 tobf(float f) { return (__bf16)f; }

// B-matrix (32x16 bf16): col = lane%16, k = (e>=8?16:0) + grp*8 + (e&7),
// matching the A fragment layout.
__device__ __forceinline__ int kOffB(int e, int grp) {
    return ((e & 8) << 1) + (grp << 3) + (e & 7);
}

// Load an A fragment row (row base pre-offset) from LDS: two contiguous
// 16-byte chunks matching the documented 16-bit A layout
// (k = (e>=8?16:0) + grp*8 + (e&7)).
__device__ __forceinline__ v16bf loadA(const __bf16* __restrict__ row, int kb, int grp) {
    union { v16bf v; uint4 q[2]; } u;
    u.q[0] = *(const uint4*)(row + kb + grp * 8);
    u.q[1] = *(const uint4*)(row + kb + 16 + grp * 8);
    return u.v;
}

// ---------------------------------------------------------------------------
// Async-stage the gathered 16 x 2048 f32 X tile into bf16 LDS tile `xs`,
// using GLOBAL_LOAD_ASYNC_TO_LDS_B128 through a 16 x 256 f32 staging buffer.
// Generic (flat) pointers to LDS carry the LDS byte offset in their low 32
// bits (ISA aperture rule), so the low dword is a valid async VDST address.
// ---------------------------------------------------------------------------
__device__ __forceinline__ void stage_x_async(
    const float* __restrict__ x, const int* __restrict__ tok,
    __bf16* __restrict__ xs, float* __restrict__ stage, int tid)
{
    for (int chunk = 0; chunk < 8; ++chunk) {
        const int c0 = chunk << 8;  // 256-column chunk
#pragma unroll
        for (int j = 0; j < 4; ++j) {
            const int i = tid * 4 + j * 1024;          // 0..4095 (f32 idx)
            const int r = i >> 8;
            const int c = (i & 255) + c0;
            const float* src = x + (size_t)tok[r] * DD + c;
            const unsigned lds = (unsigned)(size_t)(const void*)(stage + i);
            asm volatile("global_load_async_to_lds_b128 %0, %1, off"
                         :: "v"(lds), "v"(src) : "memory");
        }
        asm volatile("s_wait_asynccnt 0x0" ::: "memory");
        __syncthreads();
        for (int i = tid; i < 4096; i += 256) {
            const int r = i >> 8;
            const int c = i & 255;
            xs[(size_t)r * DD + c0 + c] = tobf(stage[i]);
        }
        __syncthreads();
    }
}

// ---------------------------------------------------------------------------
// Fused SwiGLU FFN tile core: 16 tokens x (D=2048 -> Fc -> D=2048).
// xs: [16][2048] bf16 activations (LDS), hs: [16][Fc] bf16 intermediate (LDS).
// Wg/Wu: column-offset-applied, leading dim ldgu (row-major [2048 x ldgu]).
// Wd: row-offset-applied, leading dim 2048 (row-major [Fc x 2048]).
// out accumulated with atomics, scaled per-token by scl[m], rows tok[m].
// ---------------------------------------------------------------------------
__device__ __forceinline__ void ffn_core(
    const float* __restrict__ Wg, const float* __restrict__ Wu, int ldgu,
    const float* __restrict__ Wd,
    float* __restrict__ out, int Fc,
    const int* __restrict__ tok, const float* __restrict__ scl,
    const __bf16* __restrict__ xs, __bf16* __restrict__ hs)
{
    const int tid  = threadIdx.x;
    const int lane = tid & 31;
    const int wave = tid >> 5;
    const int m16  = lane & 15;
    const int grp  = lane >> 4;

    // ---- GEMM1: g = X*Wg, u = X*Wu ; h = silu(g)*u -> LDS -----------------
    const int nT = Fc >> 4;
    for (int t = wave; t < nT; t += 8) {
        const int col0 = t << 4;
        v8f ag = {0.f,0.f,0.f,0.f,0.f,0.f,0.f,0.f};
        v8f au = {0.f,0.f,0.f,0.f,0.f,0.f,0.f,0.f};
        const float* pg = Wg + (size_t)col0 + m16;
        const float* pu = Wu + (size_t)col0 + m16;
        for (int kb = 0; kb < DD; kb += 32) {
            v16bf a = loadA(xs + (size_t)m16 * DD, kb, grp);
            v16bf bg, bu;
#pragma unroll
            for (int e = 0; e < 16; ++e) {
                const size_t k = (size_t)(kb + kOffB(e, grp));
                bg[e] = tobf(pg[k * (size_t)ldgu]);
                bu[e] = tobf(pu[k * (size_t)ldgu]);
            }
            if (kb + 32 < DD) {
                __builtin_prefetch(pg + (size_t)(kb + 32) * (size_t)ldgu, 0, 1);
                __builtin_prefetch(pu + (size_t)(kb + 32) * (size_t)ldgu, 0, 1);
            }
            ag = __builtin_amdgcn_wmma_f32_16x16x32_bf16(false, a, false, bg,
                                                         (short)0, ag, false, false);
            au = __builtin_amdgcn_wmma_f32_16x16x32_bf16(false, a, false, bu,
                                                         (short)0, au, false, false);
        }
#pragma unroll
        for (int r = 0; r < 8; ++r) {
            const float g = ag[r];
            const float u = au[r];
            const float h = (g / (1.f + __expf(-g))) * u;   // silu(g)*u
            const int m = r + (grp << 3);
            hs[(size_t)m * Fc + col0 + m16] = tobf(h);
        }
    }
    __syncthreads();

    // ---- GEMM2: y = H * Wd ; out[tok[m], :] += scl[m] * y ------------------
    for (int dt = wave; dt < (DD >> 4); dt += 8) {
        const int dcol0 = dt << 4;
        v8f acc = {0.f,0.f,0.f,0.f,0.f,0.f,0.f,0.f};
        const float* pd = Wd + (size_t)dcol0 + m16;
        for (int kb = 0; kb < Fc; kb += 32) {
            v16bf a = loadA(hs + (size_t)m16 * Fc, kb, grp);
            v16bf b;
#pragma unroll
            for (int e = 0; e < 16; ++e) {
                const size_t k = (size_t)(kb + kOffB(e, grp));
                b[e] = tobf(pd[k * (size_t)DD]);
            }
            if (kb + 32 < Fc)
                __builtin_prefetch(pd + (size_t)(kb + 32) * (size_t)DD, 0, 1);
            acc = __builtin_amdgcn_wmma_f32_16x16x32_bf16(false, a, false, b,
                                                          (short)0, acc, false, false);
        }
#pragma unroll
        for (int r = 0; r < 8; ++r) {
            const int m = r + (grp << 3);
            unsafeAtomicAdd(&out[(size_t)tok[m] * DD + dcol0 + m16],
                            scl[m] * acc[r]);
        }
    }
}

// ---------------------------------------------------------------------------
// init: zero final-output region and per-expert counters
// ---------------------------------------------------------------------------
__global__ __launch_bounds__(256) void init_kernel(float* __restrict__ outf,
                                                   int* __restrict__ counts) {
    const size_t i = (size_t)blockIdx.x * 256 + threadIdx.x;
    if (i < OUT_FINAL) outf[i] = 0.f;
    if (i < EE) counts[i] = 0;
}

// ---------------------------------------------------------------------------
// router: logits, softmax top-4 scatter, shared-expert sigmoid gate
// one block (256 threads) per token
// ---------------------------------------------------------------------------
__global__ __launch_bounds__(256) void router_kernel(
    const float* __restrict__ x, const float* __restrict__ gw,
    const float* __restrict__ sgw,
    float* __restrict__ logits_out,
    int* __restrict__ counts, int* __restrict__ idxlist,
    float* __restrict__ wlist, float* __restrict__ sg)
{
    const int t   = blockIdx.x;
    const int tid = threadIdx.x;
    const float* xr = x + (size_t)t * DD;

    float acc[17];
#pragma unroll
    for (int e = 0; e < 17; ++e) acc[e] = 0.f;

    for (int d = tid; d < DD; d += 256) {
        const float xv = xr[d];
        const float* g = gw + (size_t)d * EE;
#pragma unroll
        for (int e = 0; e < EE; ++e) acc[e] += xv * g[e];
        acc[16] += xv * sgw[d];
    }

    __shared__ float red[256][17];
#pragma unroll
    for (int e = 0; e < 17; ++e) red[tid][e] = acc[e];
    __syncthreads();
    for (int s = 128; s > 0; s >>= 1) {
        if (tid < s) {
#pragma unroll
            for (int e = 0; e < 17; ++e) red[tid][e] += red[tid + s][e];
        }
        __syncthreads();
    }

    if (tid == 0) {
        float v[17];
#pragma unroll
        for (int e = 0; e < 17; ++e) v[e] = red[0][e];

        // raw logits out
        float* lo = logits_out + (size_t)t * EE;
#pragma unroll
        for (int e = 0; e < EE; ++e) lo[e] = v[e];

        // softmax (no renormalization of top-k, per reference)
        float mx = v[0];
#pragma unroll
        for (int e = 1; e < EE; ++e) mx = fmaxf(mx, v[e]);
        float p[EE]; float sum = 0.f;
#pragma unroll
        for (int e = 0; e < EE; ++e) { p[e] = __expf(v[e] - mx); sum += p[e]; }
        const float inv = 1.f / sum;

        // top-4, lowest index wins ties (matches jax.lax.top_k)
        unsigned used = 0u;
        for (int k = 0; k < KK; ++k) {
            int best = -1; float bv = -1.f;
            for (int e = 0; e < EE; ++e) {
                if (!((used >> e) & 1u) && p[e] > bv) { bv = p[e]; best = e; }
            }
            used |= (1u << best);
            const int pos = atomicAdd(&counts[best], 1);
            idxlist[best * NTOK + pos] = t;
            wlist[best * NTOK + pos]   = bv * inv;
        }
        sg[t] = 1.f / (1.f + __expf(-v[16]));
    }
}

// LDS layout offsets (bytes)
#define SM_TOK   0
#define SM_SCL   64
#define SM_XS    128
#define SM_STAGE (128 + 16 * DD * 2)              // after xs (bf16)
#define SM_HS    (SM_STAGE + 16 * 256 * 4)        // after f32 staging buffer

// ---------------------------------------------------------------------------
// routed experts: grid.x = expert, grid.y = 16-token tile within expert list
// ---------------------------------------------------------------------------
__global__ __launch_bounds__(256) void moe_ffn_kernel(
    const float* __restrict__ x,
    const float* __restrict__ Wg, const float* __restrict__ Wu,
    const float* __restrict__ Wd,
    const int* __restrict__ counts, const int* __restrict__ idxlist,
    const float* __restrict__ wlist,
    float* __restrict__ out)
{
    const int e    = blockIdx.x;
    const int tile = blockIdx.y;
    const int cnt  = counts[e];
    if (tile * 16 >= cnt) return;

    extern __shared__ char smem[];
    int*    tok   = (int*)(smem + SM_TOK);
    float*  scl   = (float*)(smem + SM_SCL);
    __bf16* xs    = (__bf16*)(smem + SM_XS);
    float*  stage = (float*)(smem + SM_STAGE);
    __bf16* hs    = (__bf16*)(smem + SM_HS);

    const int tid = threadIdx.x;
    if (tid < 16) {
        const int slot = tile * 16 + tid;
        if (slot < cnt) {
            tok[tid] = idxlist[e * NTOK + slot];
            scl[tid] = wlist[e * NTOK + slot];
        } else {
            tok[tid] = 0;
            scl[tid] = 0.f;
        }
    }
    __syncthreads();

    stage_x_async(x, tok, xs, stage, tid);

    const float* Wge = Wg + (size_t)e * DD * FF;
    const float* Wue = Wu + (size_t)e * DD * FF;
    const float* Wde = Wd + (size_t)e * FF * DD;
    ffn_core(Wge, Wue, FF, Wde, out, FF, tok, scl, xs, hs);
}

// ---------------------------------------------------------------------------
// shared expert: grid.x = 16-token tile (contiguous), grid.y = FS chunk (704)
// sigmoid-gate is linear over the chunked down-proj partials.
// ---------------------------------------------------------------------------
#define FCHUNK 704
__global__ __launch_bounds__(256) void shared_ffn_kernel(
    const float* __restrict__ x,
    const float* __restrict__ sWg, const float* __restrict__ sWu,
    const float* __restrict__ sWd,
    const float* __restrict__ sg,
    float* __restrict__ out)
{
    const int tile  = blockIdx.x;
    const int chunk = blockIdx.y;

    extern __shared__ char smem[];
    int*    tok   = (int*)(smem + SM_TOK);
    float*  scl   = (float*)(smem + SM_SCL);
    __bf16* xs    = (__bf16*)(smem + SM_XS);
    float*  stage = (float*)(smem + SM_STAGE);
    __bf16* hs    = (__bf16*)(smem + SM_HS);

    const int tid = threadIdx.x;
    if (tid < 16) {
        const int t = tile * 16 + tid;
        tok[tid] = t;
        scl[tid] = sg[t];
    }
    __syncthreads();

    stage_x_async(x, tok, xs, stage, tid);

    const float* Wgc = sWg + (size_t)chunk * FCHUNK;         // column offset
    const float* Wuc = sWu + (size_t)chunk * FCHUNK;
    const float* Wdc = sWd + (size_t)chunk * FCHUNK * DD;    // row offset
    ffn_core(Wgc, Wuc, FSS, Wdc, out, FCHUNK, tok, scl, xs, hs);
}

// ---------------------------------------------------------------------------
extern "C" void kernel_launch(void* const* d_in, const int* in_sizes, int n_in,
                              void* d_out, int out_size, void* d_ws, size_t ws_size,
                              hipStream_t stream) {
    const float* x    = (const float*)d_in[0];
    const float* gw   = (const float*)d_in[1];
    const float* Wg   = (const float*)d_in[2];
    const float* Wu   = (const float*)d_in[3];
    const float* Wd   = (const float*)d_in[4];
    const float* sWg  = (const float*)d_in[5];
    const float* sWu  = (const float*)d_in[6];
    const float* sWd  = (const float*)d_in[7];
    const float* sgw  = (const float*)d_in[8];

    float* out    = (float*)d_out;
    float* logits = out + OUT_FINAL;

    char*  ws      = (char*)d_ws;
    int*   counts  = (int*)ws;                         // 16
    int*   idxlist = counts + EE;                      // 16*2048
    float* wlist   = (float*)(idxlist + EE * NTOK);    // 16*2048
    float* sg      = wlist + EE * NTOK;                // 2048

    // 1) zero final-output region + counters
    init_kernel<<<(int)((OUT_FINAL + 255) / 256), 256, 0, stream>>>(out, counts);

    // 2) router: logits, top-4 scatter lists, shared gate
    router_kernel<<<NTOK, 256, 0, stream>>>(x, gw, sgw, logits,
                                            counts, idxlist, wlist, sg);

    // 3) routed experts
    const size_t lds_moe = SM_HS + (size_t)16 * FF * 2;
    moe_ffn_kernel<<<dim3(EE, NTOK / 16), 256, lds_moe, stream>>>(
        x, Wg, Wu, Wd, counts, idxlist, wlist, out);

    // 4) shared expert
    const size_t lds_sh = SM_HS + (size_t)16 * FCHUNK * 2;
    shared_ffn_kernel<<<dim3(NTOK / 16, FSS / FCHUNK), 256, lds_sh, stream>>>(
        x, sWg, sWu, sWd, sg, out);
}
